// AssociativeLayerWrapper_44865228374090
// MI455X (gfx1250) — compile-verified, matrix operand order checked by reading, not run on hardware
//
#include <hip/hip_runtime.h>
#include <hip/hip_bf16.h>
#include <math.h>

// ---------------------------------------------------------------------------
// AssociativeLayer (DPFP linear-attention memory) for MI455X / gfx1250.
// f16 WMMA (f32 accumulate) everywhere; operands pre-converted to f16 once;
// GEMM tiles double-buffered in LDS via global_load_async_to_lds_b128.
// ---------------------------------------------------------------------------

typedef _Float16 v16h __attribute__((ext_vector_type(16)));
typedef float    v8f  __attribute__((ext_vector_type(8)));
typedef int      v4i_t __attribute__((vector_size(16)));

#define B_    4
#define S_    4096
#define DM_   2048   // d_model
#define DMEM_ 512    // d_mem
#define H_    16
#define DH_   128    // d_model / H
#define DQH_  32     // d_mem / H
#define DK_   192    // d_key / H  (2*NU*d_mem / H)
#define MT_   128    // NUM_MEM_TOKENS

#if defined(__HIP_DEVICE_COMPILE__) &&                                        \
    __has_builtin(__builtin_amdgcn_global_load_async_to_lds_b128) &&          \
    __has_builtin(__builtin_amdgcn_s_wait_asynccnt)
#define USE_ASYNC 1
#else
#define USE_ASYNC 0
#endif

static __device__ __forceinline__ v8f wmma_f16(v16h a, v16h b, v8f c) {
  // (neg_a, A, neg_b, B, c_mod, C, reuse_a, reuse_b)
  return __builtin_amdgcn_wmma_f32_16x16x32_f16(false, a, false, b, (short)0, c,
                                                false, false);
}

#if USE_ASYNC
typedef __attribute__((address_space(1))) v4i_t gv4i;
typedef __attribute__((address_space(3))) v4i_t lv4i;
static __device__ __forceinline__ void async_ld16(const void* g, void* l) {
  __builtin_amdgcn_global_load_async_to_lds_b128((gv4i*)g, (lv4i*)l, 0, 0);
}
#endif

static __device__ __forceinline__ void copy16(const void* g, void* l) {
#if USE_ASYNC
  async_ld16(g, l);
#else
  *(v4i_t*)l = *(const v4i_t*)g;
#endif
}

// ---------------------------------------------------------------------------
// Tiled projection: Y[M,N] = X[M,K] @ W[N,K]^T  (X,W pre-converted f16)
// Workgroup (8 waves) computes a 128x128 tile. A/B k-slices (128x32 f16)
// double-buffered in LDS via async global->LDS copies.
// Requires M%128==0, N%128==0, K%32==0.
// ---------------------------------------------------------------------------
#define HSTR 40   // LDS row stride in halfs: 32 data + 8 pad (80B, 16B-aligned)

__global__ void __launch_bounds__(256)
proj_tile_kernel(const _Float16* __restrict__ X, const _Float16* __restrict__ W,
                 float* __restrict__ Y, int M, int N, int K) {
  __shared__ alignas(16) _Float16 As[2][128][HSTR];
  __shared__ alignas(16) _Float16 Bs[2][128][HSTR];

  int tid = threadIdx.x, lane = tid & 31, wv = tid >> 5;
  int nblk = blockIdx.x * 128, mblk = blockIdx.y * 128;
  int lo = lane & 15, kh8 = (lane >> 4) * 8, mh = (lane >> 4) * 8;
  int m0 = wv * 16;
  int KS = K / 32;

  auto load_tiles = [&](int p, int k0) {
#pragma unroll
    for (int it = 0; it < 2; ++it) {
      int idx = tid + 256 * it;              // 512 segments: 128 rows x 4
      int row = idx >> 2, seg = idx & 3;     // seg = 16B chunk (8 halfs)
      copy16(X + (long)(mblk + row) * (long)K + k0 + seg * 8,
             &As[p][row][seg * 8]);
      copy16(W + (long)(nblk + row) * (long)K + k0 + seg * 8,
             &Bs[p][row][seg * 8]);
    }
  };

  v8f acc[8] = {};

  load_tiles(0, 0);
#if USE_ASYNC
  __builtin_amdgcn_s_wait_asynccnt(0);
#endif
  __syncthreads();

  int p = 0;
  for (int ks = 0; ks < KS; ++ks) {
    if (ks + 1 < KS) load_tiles(p ^ 1, (ks + 1) * 32);   // prefetch next slice

    v16h a;
#pragma unroll
    for (int i = 0; i < 8; ++i) {
      a[i]     = As[p][m0 + lo][kh8 + i];
      a[8 + i] = As[p][m0 + lo][16 + kh8 + i];
    }
#pragma unroll
    for (int nt = 0; nt < 8; ++nt) {
      v16h bfr;
#pragma unroll
      for (int i = 0; i < 8; ++i) {
        bfr[i]     = Bs[p][nt * 16 + lo][kh8 + i];
        bfr[8 + i] = Bs[p][nt * 16 + lo][16 + kh8 + i];
      }
      acc[nt] = wmma_f16(a, bfr, acc[nt]);
    }

#if USE_ASYNC
    __builtin_amdgcn_s_wait_asynccnt(0);
#endif
    __syncthreads();
    p ^= 1;
  }

#pragma unroll
  for (int nt = 0; nt < 8; ++nt)
#pragma unroll
    for (int r = 0; r < 8; ++r)
      Y[(long)(mblk + m0 + mh + r) * (long)N + nblk + nt * 16 + lo] = acc[nt][r];
}

// ---------------------------------------------------------------------------
// Small projection (N=16, used for mb): one wave per 16x16 tile,
// bias + sigmoid, f16 inputs (compact mem rows).
// ---------------------------------------------------------------------------
__global__ void __launch_bounds__(32)
proj_small_kernel(const _Float16* __restrict__ X, const _Float16* __restrict__ W,
                  const float* __restrict__ bias, float* __restrict__ Y,
                  int N, int K) {
  int lane = threadIdx.x;
  int n0 = blockIdx.x * 16;
  int m0 = blockIdx.y * 16;
  int lo = lane & 15;
  int kh8 = (lane >> 4) * 8;

  const _Float16* ap = X + (long)(m0 + lo) * (long)K;
  const _Float16* bp = W + (long)(n0 + lo) * (long)K;

  v8f acc = {};
  for (int k0 = 0; k0 < K; k0 += 32) {
    v16h a, bfr;
#pragma unroll
    for (int i = 0; i < 8; ++i) {
      a[i]       = ap[k0 + kh8 + i];
      a[8 + i]   = ap[k0 + 16 + kh8 + i];
      bfr[i]     = bp[k0 + kh8 + i];
      bfr[8 + i] = bp[k0 + 16 + kh8 + i];
    }
    if (k0 + 64 < K) {
      __builtin_prefetch(ap + k0 + 64, 0, 3);
      __builtin_prefetch(bp + k0 + 64, 0, 3);
    }
    acc = wmma_f16(a, bfr, acc);
  }

  int mh = (lane >> 4) * 8;
#pragma unroll
  for (int r = 0; r < 8; ++r) {
    float v = 1.0f / (1.0f + __expf(-(acc[r] + bias[n0 + lo])));
    Y[(long)(m0 + mh + r) * (long)N + (n0 + lo)] = v;
  }
}

// ---------------------------------------------------------------------------
// Elementwise f32 -> f16 (4 elems/thread)
// ---------------------------------------------------------------------------
__global__ void __launch_bounds__(256)
cvt_f16_kernel(const float* __restrict__ src, _Float16* __restrict__ dst,
               long n) {
  long i = ((long)blockIdx.x * 256 + threadIdx.x) * 4;
  if (i + 3 < n) {
    float4 v = *(const float4*)(src + i);
    dst[i]     = (_Float16)v.x;
    dst[i + 1] = (_Float16)v.y;
    dst[i + 2] = (_Float16)v.z;
    dst[i + 3] = (_Float16)v.w;
  }
}

// W_mem (b,h,192,128) f32 -> transposed f16 WmT (b,h,128,192): WmT[n][k]=W[k][n]
__global__ void __launch_bounds__(256)
cvt_wmemT_kernel(const float* __restrict__ W, _Float16* __restrict__ WT) {
  int bh = blockIdx.x;
  const float* s = W + (long)bh * DK_ * DH_;
  _Float16* d    = WT + (long)bh * DK_ * DH_;
  for (int i = threadIdx.x; i < DK_ * DH_; i += 256) {
    int k = i / DH_, n = i % DH_;
    d[n * DK_ + k] = (_Float16)s[i];
  }
}

// last-128-token rows of out (f32, strided) -> compact f16 (512 x 2048)
__global__ void __launch_bounds__(256)
cvt_mem_kernel(const float* __restrict__ out, _Float16* __restrict__ mem16) {
  long i = (long)blockIdx.x * 256 + threadIdx.x;   // over 512*2048
  int row = (int)(i >> 11), c = (int)(i & 2047);
  int bb = row >> 7, tok = (S_ - MT_) + (row & (MT_ - 1));
  mem16[i] = (_Float16)out[((long)(bb * S_ + tok)) * DM_ + c];
}

// ---------------------------------------------------------------------------
// read + residual: per (b,h,16-token tile)
//   mq = l2norm(dpfp(q_head))            (lanes 0..15, into LDS as f16)
//   num = mq @ W_mem[b,h]  (WMMA, B-frags from transposed f16 WmT)
//   out = num / (z.mq + eps) + hs
// ---------------------------------------------------------------------------
__global__ void __launch_bounds__(32)
read_out_kernel(const float* __restrict__ hs, const float* __restrict__ q,
                const _Float16* __restrict__ WmemT, const float* __restrict__ z,
                float* __restrict__ out) {
  __shared__ alignas(16) _Float16 mq[16][DK_ + 8];
  __shared__ float denom[16];

  int lane = threadIdx.x;
  int t0 = blockIdx.x * 16;
  int h  = blockIdx.y;
  int b  = blockIdx.z;

  if (lane < 16) {
    int t = t0 + lane;
    const float* qp = q + ((long)(b * S_ + t)) * DMEM_ + h * DQH_;
    float x[64];
#pragma unroll
    for (int i = 0; i < 32; ++i) {
      float v = qp[i];
      x[i]      = v > 0.f ? v : 0.f;
      x[32 + i] = v < 0.f ? -v : 0.f;
    }
    float ss = 0.f;
#pragma unroll
    for (int idx = 0; idx < DK_; ++idx) {          // dpfp: rep * rolled
      int jr = (idx >> 6) + 1, pos = idx & 63;
      float v = x[pos] * x[(pos - jr) & 63];
      ss += v * v;
    }
    float inv = 1.0f / fmaxf(sqrtf(ss), 1e-12f);
    const float* zp = z + ((long)(b * H_ + h)) * DK_;
    float dn = 0.f;
#pragma unroll
    for (int idx = 0; idx < DK_; ++idx) {
      int jr = (idx >> 6) + 1, pos = idx & 63;
      float v = x[pos] * x[(pos - jr) & 63] * inv;
      mq[lane][idx] = (_Float16)v;
      dn += zp[idx] * v;
    }
    denom[lane] = dn + 1e-5f;
  }
  __syncthreads();

  const _Float16* Wm = WmemT + ((long)(b * H_ + h)) * DK_ * DH_;  // [n][k]
  int lo = lane & 15, kh8 = (lane >> 4) * 8, mh = (lane >> 4) * 8;
#pragma unroll
  for (int nt = 0; nt < 8; ++nt) {
    int n0 = nt * 16;
    const _Float16* wrow = Wm + (long)(n0 + lo) * DK_;
    v8f acc = {};
#pragma unroll
    for (int ks = 0; ks < 6; ++ks) {
      int k0 = ks * 32;
      v16h a, bfr;
#pragma unroll
      for (int i = 0; i < 8; ++i) {
        a[i]       = mq[lo][k0 + kh8 + i];
        a[8 + i]   = mq[lo][k0 + 16 + kh8 + i];
        bfr[i]     = wrow[k0 + kh8 + i];
        bfr[8 + i] = wrow[k0 + 16 + kh8 + i];
      }
      acc = wmma_f16(a, bfr, acc);
    }
#pragma unroll
    for (int r = 0; r < 8; ++r) {
      int t = t0 + mh + r;
      long o = ((long)(b * S_ + t)) * DM_ + h * DH_ + n0 + lo;
      out[o] = acc[r] / denom[mh + r] + hs[o];
    }
  }
}

// ---------------------------------------------------------------------------
// Memory update: one workgroup (8 waves) per (b,h).
//   mk = l2norm(dpfp(k_head))  -> LDS (128x192 f16)
//   num2 = mk @ W_mem; prev_mv = num2/denom2; mvmbT[t][j] = (mv*mb) transposed
//   assoc = mk^T @ mvmb (192x128); W_out = W_mem + assoc
//   z_out = z + sum_j coef_j * mk_j
// ---------------------------------------------------------------------------
__global__ void __launch_bounds__(256)
update_kernel(const float* __restrict__ kbuf, const float* __restrict__ mvbuf,
              const float* __restrict__ mbbuf, const float* __restrict__ Wmem,
              const _Float16* __restrict__ WmemT, const float* __restrict__ z,
              _Float16* __restrict__ mvmbTws, float* __restrict__ Wout,
              float* __restrict__ zout) {
  __shared__ alignas(16) _Float16 mk[MT_][DK_ + 8];
  __shared__ float denom2[MT_], coef[MT_], zsh[DK_];

  int h = blockIdx.x, b = blockIdx.y;
  int tid = threadIdx.x;
  int lane = tid & 31, wv = tid >> 5;

  for (int i = tid; i < DK_; i += 256)
    zsh[i] = z[((long)(b * H_ + h)) * DK_ + i];
  __syncthreads();

  if (tid < MT_) {
    int j = tid;
    const float* kp = kbuf + ((long)(b * MT_ + j)) * DMEM_ + h * DQH_;
    float x[64];
#pragma unroll
    for (int i = 0; i < 32; ++i) {
      float v = kp[i];
      x[i]      = v > 0.f ? v : 0.f;
      x[32 + i] = v < 0.f ? -v : 0.f;
    }
    float ss = 0.f;
#pragma unroll
    for (int idx = 0; idx < DK_; ++idx) {
      int jr = (idx >> 6) + 1, pos = idx & 63;
      float v = x[pos] * x[(pos - jr) & 63];
      ss += v * v;
    }
    float inv = 1.0f / fmaxf(sqrtf(ss), 1e-12f);
    float dn = 0.f, s2 = 0.f;
#pragma unroll
    for (int idx = 0; idx < DK_; ++idx) {
      int jr = (idx >> 6) + 1, pos = idx & 63;
      float v = x[pos] * x[(pos - jr) & 63] * inv;
      mk[j][idx] = (_Float16)v;
      dn += zsh[idx] * v;
      s2 += v * v;
    }
    float d2 = dn + 1e-5f;
    denom2[j] = d2;
    float c = 1.0f - d2 / s2;
    coef[j] = fminf(fmaxf(c, 0.f), 1.f);
  }
  __syncthreads();

  const _Float16* Wm = WmemT + ((long)(b * H_ + h)) * DK_ * DH_;  // [n][k]
  _Float16* mvmbT    = mvmbTws + ((long)(b * H_ + h)) * MT_ * DH_; // [t][j]
  int lo = lane & 15, kh8 = (lane >> 4) * 8, mh = (lane >> 4) * 8;

  // num2 -> mvmbT ; wave wv owns token tile [wv*16, wv*16+16)
  {
    int m0 = wv * 16;
#pragma unroll
    for (int nt = 0; nt < 8; ++nt) {
      int n0 = nt * 16;
      const _Float16* wrow = Wm + (long)(n0 + lo) * DK_;
      v8f acc = {};
#pragma unroll
      for (int ks = 0; ks < 6; ++ks) {
        int k0 = ks * 32;
        v16h a, bfr;
#pragma unroll
        for (int i = 0; i < 8; ++i) {
          a[i]       = mk[m0 + lo][k0 + kh8 + i];
          a[8 + i]   = mk[m0 + lo][k0 + 16 + kh8 + i];
          bfr[i]     = wrow[k0 + kh8 + i];
          bfr[8 + i] = wrow[k0 + 16 + kh8 + i];
        }
        acc = wmma_f16(a, bfr, acc);
      }
#pragma unroll
      for (int r = 0; r < 8; ++r) {
        int j = m0 + mh + r;
        float prev = acc[r] / denom2[j];
        float nv  = mvbuf[((long)(b * MT_ + j)) * DM_ + h * DH_ + n0 + lo];
        float mbv = mbbuf[((long)(b * MT_ + j)) * H_ + h];
        mvmbT[(long)(n0 + lo) * MT_ + j] = (_Float16)((nv - prev) * mbv);
      }
    }
  }
  __threadfence();
  __syncthreads();

  // assoc = mk^T @ mvmb : 12 M-tiles (192) x 8 N-tiles (128), K = 128 tokens
  for (int job = wv; job < 96; job += 8) {
    int mt = job >> 3, nt = job & 7;
    int m0 = mt * 16, n0 = nt * 16;
    const _Float16* brow = mvmbT + (long)(n0 + lo) * MT_;
    v8f acc = {};
#pragma unroll
    for (int ks = 0; ks < 4; ++ks) {
      int k0 = ks * 32;
      v16h a, bfr;
#pragma unroll
      for (int i = 0; i < 8; ++i) {
        a[i]       = mk[k0 + kh8 + i][m0 + lo];        // transposed LDS read
        a[8 + i]   = mk[k0 + 16 + kh8 + i][m0 + lo];
        bfr[i]     = brow[k0 + kh8 + i];
        bfr[8 + i] = brow[k0 + 16 + kh8 + i];
      }
      acc = wmma_f16(a, bfr, acc);
    }
#pragma unroll
    for (int r = 0; r < 8; ++r) {
      long o = ((long)(b * H_ + h)) * DK_ * DH_ + (long)(m0 + mh + r) * DH_ + n0 + lo;
      Wout[o] = Wmem[o] + acc[r];
    }
  }
  __syncthreads();

  if (tid < DK_) {
    float acc = zsh[tid];
#pragma unroll 4
    for (int j = 0; j < MT_; ++j) acc += coef[j] * (float)mk[j][tid];
    zout[((long)(b * H_ + h)) * DK_ + tid] = acc;
  }
}

// ---------------------------------------------------------------------------
extern "C" void kernel_launch(void* const* d_in, const int* in_sizes, int n_in,
                              void* d_out, int out_size, void* d_ws, size_t ws_size,
                              hipStream_t stream) {
  const float* hs    = (const float*)d_in[0];
  const float* W_mq  = (const float*)d_in[1];
  const float* W_mk  = (const float*)d_in[2];
  const float* W_mv  = (const float*)d_in[3];
  const float* W_mbw = (const float*)d_in[4];
  const float* W_mbb = (const float*)d_in[5];
  const float* W_mem = (const float*)d_in[6];
  const float* z     = (const float*)d_in[7];

  float* out  = (float*)d_out;                       // (B,S,DM)
  float* Wout = out + (long)B_ * S_ * DM_;           // (B,H,DK,DH)
  float* zout = Wout + (long)B_ * H_ * DK_ * DH_;    // (B,H,DK)

  // f32 scratch
  float* ws     = (float*)d_ws;
  float* qbuf   = ws;                                // B*S*DMEM
  float* kbuf   = qbuf  + (long)B_ * S_  * DMEM_;    // B*MT*DMEM
  float* mvbuf  = kbuf  + (long)B_ * MT_ * DMEM_;    // B*MT*DM
  float* mbbuf  = mvbuf + (long)B_ * MT_ * DM_;      // B*MT*H
  // f16 scratch
  _Float16* h16    = (_Float16*)(mbbuf + (long)B_ * MT_ * H_);
  _Float16* hs16   = h16;                                      // B*S*DM
  _Float16* wmq16  = hs16  + (long)B_ * S_ * DM_;              // DMEM*DM
  _Float16* wmk16  = wmq16 + (long)DMEM_ * DM_;                // DMEM*DM
  _Float16* wmv16  = wmk16 + (long)DMEM_ * DM_;                // DM*DM
  _Float16* wmbw16 = wmv16 + (long)DM_ * DM_;                  // H*DM
  _Float16* wmemT  = wmbw16 + (long)H_ * DM_;                  // B*H*DH*DK
  _Float16* mem16  = wmemT + (long)B_ * H_ * DK_ * DH_;        // B*MT*DM
  _Float16* mvmbT  = mem16 + (long)B_ * MT_ * DM_;             // B*H*DH*MT

  // ---- one-time f32 -> f16 staging ----
  long nhs = (long)B_ * S_ * DM_;
  cvt_f16_kernel<<<(int)(nhs / 1024), 256, 0, stream>>>(hs, hs16, nhs);
  long nwq = (long)DMEM_ * DM_;
  cvt_f16_kernel<<<(int)(nwq / 1024), 256, 0, stream>>>(W_mq, wmq16, nwq);
  cvt_f16_kernel<<<(int)(nwq / 1024), 256, 0, stream>>>(W_mk, wmk16, nwq);
  long nwv = (long)DM_ * DM_;
  cvt_f16_kernel<<<(int)(nwv / 1024), 256, 0, stream>>>(W_mv, wmv16, nwv);
  long nwb = (long)H_ * DM_;
  cvt_f16_kernel<<<(int)((nwb + 1023) / 1024), 256, 0, stream>>>(W_mbw, wmbw16, nwb);
  cvt_wmemT_kernel<<<B_ * H_, 256, 0, stream>>>(W_mem, wmemT);

  // q = hs @ W_mq^T   (16384x2048 @ 2048x512)
  proj_tile_kernel<<<dim3(DMEM_ / 128, (B_ * S_) / 128), 256, 0, stream>>>(
      hs16, wmq16, qbuf, B_ * S_, DMEM_, DM_);

  // out = read(mq, W_mem, z) + hs
  read_out_kernel<<<dim3(S_ / 16, H_, B_), 32, 0, stream>>>(
      hs, qbuf, wmemT, z, out);

  // compact last-128-token rows of out into f16
  cvt_mem_kernel<<<(B_ * MT_ * DM_) / 256, 256, 0, stream>>>(out, mem16);

  // mem-token projections
  proj_tile_kernel<<<dim3(DMEM_ / 128, (B_ * MT_) / 128), 256, 0, stream>>>(
      mem16, wmk16, kbuf, B_ * MT_, DMEM_, DM_);
  proj_tile_kernel<<<dim3(DM_ / 128, (B_ * MT_) / 128), 256, 0, stream>>>(
      mem16, wmv16, mvbuf, B_ * MT_, DM_, DM_);
  proj_small_kernel<<<dim3(H_ / 16, (B_ * MT_) / 16), 32, 0, stream>>>(
      mem16, wmbw16, W_mbb, mbbuf, H_, DM_);

  // associative memory update
  update_kernel<<<dim3(H_, B_), 256, 0, stream>>>(
      kbuf, mvbuf, mbbuf, W_mem, wmemT, z, mvmbT, Wout, zout);
}
